// EFAdLIF_63084479644193
// MI455X (gfx1250) — matrix operand order repaired and checked by reading, not compiled, not run on hardware
//
#include <hip/hip_runtime.h>
#include <stdint.h>

// EF-AdLIF fused forward for MI455X (gfx1250, wave32, WMMA).
// grid = 32 WGs = 8 batch-groups (16 samples, WMMA M=16) x 4 N-slices (128 neurons).
// Per step: fused FF GEMM (x@W^T) + recurrent GEMM (z@Wrec^T) as bf16 WMMA with
// ALL weight fragments resident in VGPRs (~544/wave, 1 wave/SIMD), LIF dynamics
// in the C-matrix register layout. z exchanged between the 4 slice-WGs of a group
// through a double-buffered global buffer (async-loaded into LDS) with a
// monotonic atomic counter barrier.

namespace {

constexpr int TT = 500;
constexpr int BB = 128;
constexpr int KIN = 256;
constexpr int NN = 512;

constexpr int BSLICE = 16;                 // WMMA M
constexpr int NSLICE = 128;                // neurons per WG
constexpr int NGROUPS = BB / BSLICE;       // 8
constexpr int NSL = NN / NSLICE;           // 4
constexpr int NBLOCKS = NGROUPS * NSL;     // 32
constexpr int NTHREADS = 128;              // 4 waves

constexpr int XSTRIDE = KIN + 8;           // 264 (bank-conflict pad)
constexpr int ZSTRIDE = NN + 8;            // 520
constexpr int KSTEPS_FF = KIN / 32;        // 8
constexpr int KSTEPS_RC = NN / 32;         // 16

typedef __bf16 v16bf __attribute__((ext_vector_type(16)));
typedef float  v8f   __attribute__((ext_vector_type(8)));
typedef int    v4i   __attribute__((ext_vector_type(4)));

union Frag {
  v16bf v;
  unsigned short h[16];
  uint4 q[2];
};

__device__ __forceinline__ unsigned short f2bf(float f) {
  union { float f; uint32_t u; } x; x.f = f;
  uint32_t r = ((x.u >> 16) & 1u) + 0x7FFFu;          // round-to-nearest-even
  return (unsigned short)((x.u + r) >> 16);
}

__device__ __forceinline__ void pack16(Frag& fr, const float* p) {
  const float4* q = (const float4*)p;
  float4 f0 = q[0], f1 = q[1], f2 = q[2], f3 = q[3];
  fr.h[0] = f2bf(f0.x);  fr.h[1] = f2bf(f0.y);  fr.h[2] = f2bf(f0.z);  fr.h[3] = f2bf(f0.w);
  fr.h[4] = f2bf(f1.x);  fr.h[5] = f2bf(f1.y);  fr.h[6] = f2bf(f1.z);  fr.h[7] = f2bf(f1.w);
  fr.h[8] = f2bf(f2.x);  fr.h[9] = f2bf(f2.y);  fr.h[10] = f2bf(f2.z); fr.h[11] = f2bf(f2.w);
  fr.h[12] = f2bf(f3.x); fr.h[13] = f2bf(f3.y); fr.h[14] = f2bf(f3.z); fr.h[15] = f2bf(f3.w);
}

} // namespace

#if defined(__has_builtin)
#if __has_builtin(__builtin_amdgcn_global_load_async_to_lds_b128) && \
    __has_builtin(__builtin_amdgcn_s_wait_asynccnt)
#define HAVE_ASYNC_LDS 1
#endif
#endif
#ifndef HAVE_ASYNC_LDS
#define HAVE_ASYNC_LDS 0
#endif

extern "C" __global__ void adlif_init(unsigned int* cnt) {
  if (threadIdx.x < NGROUPS) cnt[threadIdx.x] = 0u;
}

extern "C" __global__ void __launch_bounds__(NTHREADS, 1)
adlif_fused(const float* __restrict__ x, const float* __restrict__ W,
            const float* __restrict__ bias, const float* __restrict__ Wrec,
            const float* __restrict__ tau_u_w, const float* __restrict__ tau_w_w,
            const float* __restrict__ av, const float* __restrict__ bv,
            float* __restrict__ out,
            unsigned int* __restrict__ cnt, unsigned short* __restrict__ zx) {
  __shared__ unsigned short lds_x[BSLICE * XSTRIDE];   //  8448 B (x_t tile, bf16)
  __shared__ unsigned short lds_z[BSLICE * ZSTRIDE];   // 16640 B (z_{t-1}, bf16)

  const int tid = threadIdx.x;
  const int wv = tid >> 5;
  const int lane = tid & 31;
  const int g = blockIdx.x >> 2;   // batch group 0..7
  const int ns = blockIdx.x & 3;   // n-slice 0..3
  const int bbase = g * BSLICE;
  const int nbase = ns * NSLICE;
  const int mrow = lane & 15;      // A-matrix row this lane supplies
  const int hi = lane >> 4;        // lane half (K sub-block select)

  // ---- per-lane column parameters (n fixed per lane per tile in C layout) ----
  int ncol[2];
  float du[2], odu[2], dw[2], wq[2], aa[2], bb[2], bs[2];
#pragma unroll
  for (int tp = 0; tp < 2; ++tp) {
    const int n = nbase + (wv * 2 + tp) * 16 + (lane & 15);
    ncol[tp] = n;
    const float tuw = tau_u_w[n], tww = tau_w_w[n];
    const float tau_u = tuw * 25.0f + (1.0f - tuw) * 5.0f;
    const float tau_w = tww * 300.0f + (1.0f - tww) * 30.0f;
    du[tp] = __expf(-1.0f / tau_u);
    odu[tp] = 1.0f - du[tp];
    dw[tp] = __expf(-1.0f / tau_w);
    wq[tp] = (1.0f - dw[tp]) * 20.0f;   // (1-decay_w)*Q
    aa[tp] = av[n];
    bb[tp] = bv[n];
    bs[tp] = bias[n];
  }

  // ---- ALL weight B-fragments resident in VGPRs for all 500 steps ----
  // B 32x16 bf16 layout: lane holds K=(hi?16:0)+j of column n=lane%16.
  Frag brec[2][KSTEPS_RC];   // Wrec: 2 x 16 x 8 = 256 VGPRs
  Frag bff[2][KSTEPS_FF];    // W   : 2 x  8 x 8 = 128 VGPRs
#pragma unroll
  for (int tp = 0; tp < 2; ++tp) {
    const float* wr = Wrec + (size_t)ncol[tp] * NN;
#pragma unroll
    for (int ks = 0; ks < KSTEPS_RC; ++ks)
      pack16(brec[tp][ks], wr + ks * 32 + hi * 16);
    const float* wf = W + (size_t)ncol[tp] * KIN;
#pragma unroll
    for (int ks = 0; ks < KSTEPS_FF; ++ks)
      pack16(bff[tp][ks], wf + ks * 32 + hi * 16);
  }

  // ---- state in C-matrix register layout: VGPR j -> (m=j+8*hi, n=ncol[tp]) ----
  float u[2][8], w[2][8], zp[2][8];
#pragma unroll
  for (int tp = 0; tp < 2; ++tp)
#pragma unroll
    for (int j = 0; j < 8; ++j) { u[tp][j] = 0.0f; w[tp][j] = 0.0f; zp[tp][j] = 0.0f; }

  for (int t = 0; t < TT; ++t) {
    // ---- stage x_t tile (16 x 256 fp32 -> bf16 LDS); prefetch x_{t+1} ----
    {
      const int r = tid >> 3;
      const int c0 = (tid & 7) * 32;
      const float* xp = x + ((size_t)t * BB + (bbase + r)) * KIN + c0;
      unsigned short* dp = lds_x + r * XSTRIDE + c0;
#pragma unroll
      for (int j = 0; j < 32; j += 4) {
        float4 f = *(const float4*)(xp + j);
        dp[j] = f2bf(f.x); dp[j + 1] = f2bf(f.y);
        dp[j + 2] = f2bf(f.z); dp[j + 3] = f2bf(f.w);
      }
      if (t + 1 < TT) __builtin_prefetch(xp + (size_t)BB * KIN, 0, 1);
    }

    // ---- inter-WG barrier: wait until all 4 slice-WGs finished step t-1 ----
    if (t > 0 && tid == 0) {
      while (__hip_atomic_load(cnt + g, __ATOMIC_ACQUIRE, __HIP_MEMORY_SCOPE_AGENT) <
             4u * (unsigned)t) {
        __builtin_amdgcn_s_sleep(1);
      }
    }
    __syncthreads();

    if (t > 0) {
      // ---- stage z_{t-1} (16 x 512 bf16 = 16 KB) exchange buffer -> LDS ----
      const unsigned short* zsrc =
          zx + ((size_t)((t - 1) & 1) * NGROUPS + g) * (BSLICE * NN);
      const int r = tid >> 3;
      const int c = (tid & 7) * 64;
      const unsigned short* gs = zsrc + r * NN + c;
      unsigned short* ld = lds_z + r * ZSTRIDE + c;
#if HAVE_ASYNC_LDS
      // async-to-LDS bypasses WGP$ and reads L2, the coherence point the
      // writer's release fence flushed to -- no reader-side cache inv needed.
#pragma unroll
      for (int j = 0; j < 8; ++j) {
        __builtin_amdgcn_global_load_async_to_lds_b128(
            (__attribute__((address_space(1))) v4i*)(gs + j * 8),
            (__attribute__((address_space(3))) v4i*)(ld + j * 8), 0, 0);
      }
      __builtin_amdgcn_s_wait_asynccnt(0);
#else
      __threadfence();  // acquire: invalidate stale lines before reading peers' z
#pragma unroll
      for (int j = 0; j < 8; ++j)
        *(uint4*)(ld + j * 8) = *(const uint4*)(gs + j * 8);
#endif
      __syncthreads();
    }

    // ---- accumulators: init with bias ----
    v8f acc[2];
#pragma unroll
    for (int tp = 0; tp < 2; ++tp)
#pragma unroll
      for (int j = 0; j < 8; ++j) acc[tp][j] = bs[tp];

    // ---- feed-forward GEMM: acc += x_t @ W^T  (8 k-steps of 32) ----
#pragma unroll
    for (int ks = 0; ks < KSTEPS_FF; ++ks) {
      Frag A;
      const int kb = ks * 32 + hi * 16;
      A.q[0] = *(const uint4*)(lds_x + mrow * XSTRIDE + kb);
      A.q[1] = *(const uint4*)(lds_x + mrow * XSTRIDE + kb + 8);
#pragma unroll
      for (int tp = 0; tp < 2; ++tp) {
        acc[tp] = __builtin_amdgcn_wmma_f32_16x16x32_bf16(
            false, A.v, false, bff[tp][ks].v, (short)0, acc[tp], false, false);
      }
    }

    // ---- recurrent GEMM: acc += z_{t-1} @ Wrec^T (16 k-steps of 32) ----
    if (t > 0) {
#pragma unroll
      for (int ks = 0; ks < KSTEPS_RC; ++ks) {
        Frag A;
        const int kb = ks * 32 + hi * 16;
        A.q[0] = *(const uint4*)(lds_z + mrow * ZSTRIDE + kb);
        A.q[1] = *(const uint4*)(lds_z + mrow * ZSTRIDE + kb + 8);
#pragma unroll
        for (int tp = 0; tp < 2; ++tp) {
          acc[tp] = __builtin_amdgcn_wmma_f32_16x16x32_bf16(
              false, A.v, false, brec[tp][ks].v, (short)0, acc[tp], false, false);
        }
      }
    }

    // ---- LIF dynamics in-register; z -> exchange buffer (bf16) first ----
    unsigned short* zdst = zx + ((size_t)(t & 1) * NGROUPS + g) * (BSLICE * NN);
#pragma unroll
    for (int tp = 0; tp < 2; ++tp) {
#pragma unroll
      for (int j = 0; j < 8; ++j) {
        const float soma = acc[tp][j];
        const float un = du[tp] * u[tp][j] + odu[tp] * (soma - w[tp][j]);
        const float sp = (un - 1.0f) > 0.0f ? 1.0f : 0.0f;
        const float wn = dw[tp] * w[tp][j] + wq[tp] * (aa[tp] * u[tp][j] + bb[tp] * zp[tp][j]);
        u[tp][j] = un * (1.0f - sp);
        w[tp][j] = wn;
        zp[tp][j] = sp;
        const int m = j + hi * 8;
        zdst[m * NN + ncol[tp]] = f2bf(sp);
      }
    }
    __threadfence();   // release z before arrival
    __syncthreads();
    if (tid == 0)
      __hip_atomic_fetch_add(cnt + g, 1u, __ATOMIC_RELEASE, __HIP_MEMORY_SCOPE_AGENT);

    // ---- d_out stores off the barrier critical path (nobody reads them) ----
#pragma unroll
    for (int tp = 0; tp < 2; ++tp) {
#pragma unroll
      for (int j = 0; j < 8; ++j) {
        const int m = j + hi * 8;
        out[((size_t)t * BB + (bbase + m)) * NN + ncol[tp]] = zp[tp][j];
      }
    }
  }
}

extern "C" void kernel_launch(void* const* d_in, const int* in_sizes, int n_in,
                              void* d_out, int out_size, void* d_ws, size_t ws_size,
                              hipStream_t stream) {
  (void)in_sizes; (void)n_in; (void)out_size; (void)ws_size;
  const float* x      = (const float*)d_in[0];
  const float* W      = (const float*)d_in[1];
  const float* bias   = (const float*)d_in[2];
  const float* Wrec   = (const float*)d_in[3];
  const float* tau_uw = (const float*)d_in[4];
  const float* tau_ww = (const float*)d_in[5];
  const float* a      = (const float*)d_in[6];
  const float* b      = (const float*)d_in[7];
  float* out = (float*)d_out;

  // ws layout: [0,256): group arrival counters; [256, 256+256KB): z exchange
  unsigned int* cnt = (unsigned int*)d_ws;
  unsigned short* zx = (unsigned short*)((char*)d_ws + 256);

  adlif_init<<<dim3(1), dim3(32), 0, stream>>>(cnt);
  adlif_fused<<<dim3(NBLOCKS), dim3(NTHREADS), 0, stream>>>(
      x, W, bias, Wrec, tau_uw, tau_ww, a, b, out, cnt, zx);
}
